// ADMDecoder_65335042506806
// MI455X (gfx1250) — matrix operand order repaired and checked by reading, not compiled, not run on hardware
//
#include <hip/hip_runtime.h>
#include <math.h>

// ---------------------------------------------------------------------------
// MI455X (gfx1250) fused bf16-WMMA implementation of the pair-attention
// decoder. All GEMMs lower to v_wmma_f32_16x16x32_bf16 (fp32 accumulate).
// Memory-bound workload: weights (~13MB) stay hot in the 192MB L2 (prefetched
// via global_prefetch_b8) and are reused by all 2048 node-workgroups; per-node
// activation slabs are staged once into LDS as bf16 and WMMA A-fragments are
// fed from LDS to avoid redundant VMEM re-reads per output tile.
// Workspace requirement: ~440 MB.
// ---------------------------------------------------------------------------

typedef __attribute__((ext_vector_type(16))) __bf16 bf16x16;
typedef __attribute__((ext_vector_type(8)))  float  f32x8;

#define NN   2048
#define KNB  32
#define EE   16
#define KTOT 48
#define DD   256
#define HH   8
#define DKH  32
#define HIDN 512
#define F3   768

__device__ __forceinline__ f32x8 zero8() {
  f32x8 z = {0.f, 0.f, 0.f, 0.f, 0.f, 0.f, 0.f, 0.f};
  return z;
}

__device__ __forceinline__ f32x8 wmma_bf16(bf16x16 a, bf16x16 b, f32x8 c) {
  // v_wmma_f32_16x16x32_bf16: (neg_a, A, neg_b, B, c_mod, C, reuse_a, reuse_b)
  return __builtin_amdgcn_wmma_f32_16x16x32_bf16(false, a, false, b, (short)0, c,
                                                 false, false);
}

// A-matrix (16x32 bf16): lane = 16*hf + m; k0 = (v>=4?16:0) + hf*8 + (v&3)*2
template <typename FA>
__device__ __forceinline__ bf16x16 mk_a(FA getA, int mbase) {
  const int lane = threadIdx.x & 31;
  const int hf = lane >> 4;
  const int m = mbase + (lane & 15);
  bf16x16 a;
#pragma unroll
  for (int v = 0; v < 8; ++v) {
    int k0 = ((v & 4) ? 16 : 0) + hf * 8 + (v & 3) * 2;
    a[2 * v]     = (__bf16)getA(m, k0);
    a[2 * v + 1] = (__bf16)getA(m, k0 + 1);
  }
  return a;
}

// B-matrix (32x16 bf16): lane = 16*hf + n; k0 = (v>=4?16:0) + hf*8 + (v&3)*2
template <typename FB>
__device__ __forceinline__ bf16x16 mk_b(FB getB, int nbase) {
  const int lane = threadIdx.x & 31;
  const int hf = lane >> 4;
  const int n = nbase + (lane & 15);
  bf16x16 b;
#pragma unroll
  for (int v = 0; v < 8; ++v) {
    int k0 = ((v & 4) ? 16 : 0) + hf * 8 + (v & 3) * 2;
    b[2 * v]     = (__bf16)getB(k0, n);
    b[2 * v + 1] = (__bf16)getB(k0 + 1, n);
  }
  return b;
}

// C/D (16x16 f32): VGPR r -> M = r + 8*hf, N = lane&15
template <typename FP>
__device__ __forceinline__ void st_c(const f32x8& c, FP put, int mbase, int nbase) {
  const int lane = threadIdx.x & 31;
  const int hf = lane >> 4;
  const int n = nbase + (lane & 15);
#pragma unroll
  for (int r = 0; r < 8; ++r) put(mbase + r + 8 * hf, n, c[r]);
}

__device__ __forceinline__ float gelu_tanh(float g) {
  return 0.5f * g * (1.f + tanhf(0.7978845608028654f * (g + 0.044715f * g * g * g)));
}

// prefetch `floats` worth of a weight matrix into cache (one line / 64 floats)
__device__ __forceinline__ void prefetch_w(const float* w, int floats, int t) {
  for (int i = t; i < floats / 64; i += 256) __builtin_prefetch(&w[i * 64], 0, 0);
}

// ---------------------------------------------------------------------------
// K1: local = LN(local + onehot@W_aa_local); pairb = [pair + W_aa_pair[idx] | extra]
// ---------------------------------------------------------------------------
__global__ void k_embed(const int* aa, const float* local, const float* pair,
                        const float* extra_pair, const int* nbr, const float* Waal,
                        const float* Waap, const float* ln0s, const float* ln0b,
                        float* localb, float* pairb) {
  int n = blockIdx.x, t = threadIdx.x;
  __shared__ float red[256];
  int a1 = aa[n];
  float x = local[n * DD + t] + Waal[a1 * DD + t];
  red[t] = x;
  __syncthreads();
  for (int s = 128; s > 0; s >>= 1) {
    if (t < s) red[t] += red[t + s];
    __syncthreads();
  }
  float mean = red[0] / DD;
  __syncthreads();
  float d = x - mean;
  red[t] = d * d;
  __syncthreads();
  for (int s = 128; s > 0; s >>= 1) {
    if (t < s) red[t] += red[t + s];
    __syncthreads();
  }
  float rstd = rsqrtf(red[0] / DD + 1e-5f);
  localb[n * DD + t] = d * rstd * ln0s[t] + ln0b[t];

  for (int r = 0; r < KTOT; ++r) {
    float v;
    if (r < KNB) {
      int nb = nbr[n * KNB + r];
      int a2 = aa[nb < 0 ? 0 : nb];
      v = pair[(n * KNB + r) * DD + t] + Waap[(a1 * 21 + a2) * DD + t];
    } else {
      v = extra_pair[(n * EE + (r - KNB)) * DD + t];
    }
    pairb[((size_t)n * KTOT + r) * DD + t] = v;
  }
}

// ---------------------------------------------------------------------------
// K_nb: nbb rows 0..31 = gather local[neighbours]; rows 32..47 = extra @ W_extra
// ---------------------------------------------------------------------------
__global__ void k_nbfeat(const float* localb, const float* pairb, const int* nbr,
                         const float* Wextra, float* nbb) {
  __shared__ __bf16 Ea[16][DD];  // 8 KB staged extra rows (A operand)
  int n = blockIdx.x, t = threadIdx.x, w = t >> 5;
  for (int j = 0; j < KNB; ++j) {
    int nb = nbr[n * KNB + j];
    nbb[((size_t)n * KTOT + j) * DD + t] = localb[(nb < 0 ? 0 : nb) * DD + t];
  }
  const float* A = pairb + ((size_t)n * KTOT + KNB) * DD;  // pre-attn extra rows
  for (int i = t; i < 16 * DD; i += 256) Ea[i / DD][i % DD] = (__bf16)A[i];
  __syncthreads();
  for (int tile = w; tile < 16; tile += 8) {
    int nb0 = tile * 16;
    f32x8 acc = zero8();
    for (int kc = 0; kc < DD; kc += 32) {
      bf16x16 a = mk_a([&](int m, int k) { return (float)Ea[m][kc + k]; }, 0);
      bf16x16 b = mk_b([&](int k, int c) { return Wextra[(kc + k) * DD + c]; }, nb0);
      acc = wmma_bf16(a, b, acc);
    }
    st_c(acc,
         [&](int m, int c, float v) { nbb[((size_t)n * KTOT + KNB + m) * DD + c] = v; },
         0, nb0);
  }
}

// ---------------------------------------------------------------------------
// K2a: Q/K/V projections (48x256x256 WMMA) with per-head LN on Q,K -> bf16 bufs
// ---------------------------------------------------------------------------
__global__ void k_qkv(const float* pairb, const float* Wq, const float* Wk,
                      const float* Wv, __bf16* qb, __bf16* kb, __bf16* vb) {
  __shared__ __bf16 Pa[KTOT][DD];  // 24 KB staged pair slab (A operand)
  __shared__ __bf16 T[KTOT][DD];   // 24 KB result staging
  int n = blockIdx.x, t = threadIdx.x, w = t >> 5;
  const float* A = pairb + (size_t)n * KTOT * DD;
  prefetch_w(Wq, DD * DD, t);
  prefetch_w(Wk, DD * DD, t);
  prefetch_w(Wv, DD * DD, t);
  for (int i = t; i < KTOT * DD; i += 256) Pa[i / DD][i % DD] = (__bf16)A[i];
  __syncthreads();
  const float* Ws[3] = {Wq, Wk, Wv};
  __bf16* dst[3] = {qb + (size_t)n * KTOT * DD, kb + (size_t)n * KTOT * DD,
                    vb + (size_t)n * KTOT * DD};
  for (int s3 = 0; s3 < 3; ++s3) {
    const float* W = Ws[s3];
    for (int tile = w; tile < 48; tile += 8) {
      int mt = tile % 3, nt = tile / 3;
      f32x8 acc = zero8();
      for (int kc = 0; kc < DD; kc += 32) {
        bf16x16 a = mk_a([&](int m, int k) { return (float)Pa[m][kc + k]; }, mt * 16);
        bf16x16 b = mk_b([&](int k, int c) { return W[(kc + k) * DD + c]; }, nt * 16);
        acc = wmma_bf16(a, b, acc);
      }
      st_c(acc, [&](int m, int c, float v) { T[m][c] = (__bf16)v; }, mt * 16, nt * 16);
    }
    __syncthreads();
    if (s3 < 2) {  // per-head LayerNorm over DK=32 (no scale/bias)
      for (int g = t; g < KTOT * HH; g += 256) {
        int row = g / HH, h = g % HH;
        float s = 0.f, ss = 0.f;
        for (int c = 0; c < DKH; ++c) {
          float x = (float)T[row][h * DKH + c];
          s += x;
          ss += x * x;
        }
        float mu = s / DKH;
        float rs = rsqrtf(ss / DKH - mu * mu + 1e-5f);
        for (int c = 0; c < DKH; ++c) {
          float x = (float)T[row][h * DKH + c];
          T[row][h * DKH + c] = (__bf16)((x - mu) * rs);
        }
      }
      __syncthreads();
    }
    __bf16* dp = dst[s3];
    for (int i = t; i < KTOT * DD; i += 256) dp[i] = T[i / DD][i % DD];
    __syncthreads();
  }
}

// ---------------------------------------------------------------------------
// K2b: per-head scores (WMMA) + masked softmax + attn@V (WMMA) + Wo projection
//      with LDS-atomic LayerNorm stats, residual-add into pairb
// ---------------------------------------------------------------------------
__global__ void k_attn(const __bf16* qb, const __bf16* kb, const __bf16* vb,
                       const float* Woa, const float* lnas, const float* lnab,
                       const int* nbr, const float* mask, const unsigned char* emask,
                       float* pairb) {
  __shared__ __bf16 Qh[KTOT][DKH], Kh[KTOT][DKH], Vh[KTOT][DKH];
  __shared__ float Sc[KTOT][KTOT];
  __shared__ __bf16 AO[KTOT][DD];
  __shared__ float maskf[KTOT];
  __shared__ float rsum[KTOT], rssq[KTOT], rmean[KTOT], rrstd[KTOT];
  int n = blockIdx.x, t = threadIdx.x, w = t >> 5;
  prefetch_w(Woa, DD * DD, t);
  if (t < KTOT) {
    bool mk;
    if (t < KNB) {
      int nb = nbr[n * KNB + t];
      mk = (nb != -1) && (mask[nb < 0 ? 0 : nb] > 0.f);
    } else {
      mk = emask[n * EE + (t - KNB)] != 0;
    }
    maskf[t] = mk ? 1.f : 0.f;
  }
  __syncthreads();
  const __bf16* Q = qb + (size_t)n * KTOT * DD;
  const __bf16* Kg = kb + (size_t)n * KTOT * DD;
  const __bf16* V = vb + (size_t)n * KTOT * DD;
  const float inv = 0.17677669529663687f;  // 1/sqrt(32)

  for (int h = 0; h < HH; ++h) {
    for (int i = t; i < KTOT * DKH; i += 256) {
      int r = i / DKH, c = i % DKH;
      Qh[r][c] = Q[r * DD + h * DKH + c];
      Kh[r][c] = Kg[r * DD + h * DKH + c];
      Vh[r][c] = V[r * DD + h * DKH + c];
    }
    __syncthreads();
    // scores = Qh @ Kh^T : 3x3 tiles, K=32 single chunk
    for (int tile = w; tile < 9; tile += 8) {
      int mt = tile % 3, nt = tile / 3;
      bf16x16 a = mk_a([&](int m, int k) { return (float)Qh[m][k]; }, mt * 16);
      bf16x16 b = mk_b([&](int k, int c) { return (float)Kh[c][k]; }, nt * 16);
      f32x8 acc = wmma_bf16(a, b, zero8());
      st_c(acc,
           [&](int m, int c, float v) {
             Sc[m][c] = (maskf[m] > 0.f && maskf[c] > 0.f) ? v * inv : -1e9f;
           },
           mt * 16, nt * 16);
    }
    __syncthreads();
    if (t < KTOT) {  // softmax over k
      float mx = -3.4e38f;
      for (int c = 0; c < KTOT; ++c) mx = fmaxf(mx, Sc[t][c]);
      float se = 0.f;
      for (int c = 0; c < KTOT; ++c) {
        float e = __expf(Sc[t][c] - mx);
        Sc[t][c] = e;
        se += e;
      }
      float r = 1.f / se;
      for (int c = 0; c < KTOT; ++c) Sc[t][c] *= r;
    }
    __syncthreads();
    // out_h = attn @ Vh : 3x2 tiles, K padded 48->64
    for (int tile = w; tile < 6; tile += 8) {
      int mt = tile % 3, nt = tile / 3;
      f32x8 acc = zero8();
      for (int kc = 0; kc < 64; kc += 32) {
        bf16x16 a = mk_a(
            [&](int m, int k) { int kk = kc + k; return kk < KTOT ? Sc[m][kk] : 0.f; },
            mt * 16);
        bf16x16 b = mk_b(
            [&](int k, int c) {
              int kk = kc + k;
              return kk < KTOT ? (float)Vh[kk][c] : 0.f;
            },
            nt * 16);
        acc = wmma_bf16(a, b, acc);
      }
      st_c(acc, [&](int m, int c, float v) { AO[m][h * DKH + c] = (__bf16)v; }, mt * 16,
           nt * 16);
    }
    __syncthreads();
  }

  if (t < KTOT) { rsum[t] = 0.f; rssq[t] = 0.f; }
  __syncthreads();
  // proj = AO @ Wo_attn, LN stats via LDS atomics, then LN + residual
  f32x8 accs[6];
#pragma unroll
  for (int i6 = 0; i6 < 6; ++i6) {
    int tile = w + i6 * 8;
    int mt = tile % 3, nt = tile / 3;
    f32x8 acc = zero8();
    for (int kc = 0; kc < DD; kc += 32) {
      bf16x16 a = mk_a([&](int m, int k) { return (float)AO[m][kc + k]; }, mt * 16);
      bf16x16 b = mk_b([&](int k, int c) { return Woa[(kc + k) * DD + c]; }, nt * 16);
      acc = wmma_bf16(a, b, acc);
    }
    accs[i6] = acc;
    int lane = t & 31, hf = lane >> 4;
#pragma unroll
    for (int r = 0; r < 8; ++r) {
      int m = mt * 16 + r + 8 * hf;
      atomicAdd(&rsum[m], acc[r]);
      atomicAdd(&rssq[m], acc[r] * acc[r]);
    }
  }
  __syncthreads();
  if (t < KTOT) {
    float mu = rsum[t] / DD;
    rmean[t] = mu;
    rrstd[t] = rsqrtf(rssq[t] / DD - mu * mu + 1e-5f);
  }
  __syncthreads();
  float* Pp = pairb + (size_t)n * KTOT * DD;
#pragma unroll
  for (int i6 = 0; i6 < 6; ++i6) {
    int tile = w + i6 * 8;
    int mt = tile % 3, nt = tile / 3;
    st_c(accs[i6],
         [&](int m, int c, float v) {
           float y = (v - rmean[m]) * rrstd[m] * lnas[c] + lnab[c];
           Pp[m * DD + c] += y;
         },
         mt * 16, nt * 16);
  }
}

// ---------------------------------------------------------------------------
// K3: pair gMLP: feats=[pair|center|nb] (768) -> gated 512 -> 256, LN, residual.
// Feats are staged per 16-row Mtile in two 16x384 bf16 slabs; gated accumulators
// stay register-resident across the K halves.
// ---------------------------------------------------------------------------
__global__ void k_gmlp_pair(float* pairb, const float* localb, const float* nbb,
                            const float* Wg, const float* Wv, const float* Wo,
                            const float* lnps, const float* lnpb, float* fpub) {
  __shared__ __bf16 Ft[16][384];     // 12 KB staged feats slab
  __shared__ __bf16 Hb[KTOT][HIDN];  // 48 KB hidden
  __shared__ float rsum[KTOT], rssq[KTOT], rmean[KTOT], rrstd[KTOT];
  int n = blockIdx.x, t = threadIdx.x, w = t >> 5;
  prefetch_w(Wg, F3 * HIDN, t);
  prefetch_w(Wv, F3 * HIDN, t);
  float* Pp = pairb + (size_t)n * KTOT * DD;
  const float* Lc = localb + (size_t)n * DD;
  const float* Nb = nbb + (size_t)n * KTOT * DD;
  auto getF = [&](int m, int k) -> float {
    if (k < DD) return Pp[m * DD + k];
    if (k < 2 * DD) return Lc[k - DD];
    return Nb[m * DD + (k - 2 * DD)];
  };
  for (int mt = 0; mt < 3; ++mt) {
    f32x8 ag[4], av[4];
#pragma unroll
    for (int i = 0; i < 4; ++i) { ag[i] = zero8(); av[i] = zero8(); }
    for (int hlf = 0; hlf < 2; ++hlf) {
      for (int i = t; i < 16 * 384; i += 256) {
        int r = i / 384, kk = i % 384;
        Ft[r][kk] = (__bf16)getF(mt * 16 + r, hlf * 384 + kk);
      }
      __syncthreads();
#pragma unroll
      for (int i4 = 0; i4 < 4; ++i4) {
        int nt = w + i4 * 8;
        for (int kc = 0; kc < 384; kc += 32) {
          bf16x16 a = mk_a([&](int m, int k) { return (float)Ft[m][kc + k]; }, 0);
          int kg = hlf * 384 + kc;
          bf16x16 bg =
              mk_b([&](int k, int c) { return Wg[(kg + k) * HIDN + c]; }, nt * 16);
          bf16x16 bv =
              mk_b([&](int k, int c) { return Wv[(kg + k) * HIDN + c]; }, nt * 16);
          ag[i4] = wmma_bf16(a, bg, ag[i4]);
          av[i4] = wmma_bf16(a, bv, av[i4]);
        }
      }
      __syncthreads();
    }
    int lane = t & 31, hf = lane >> 4;
#pragma unroll
    for (int i4 = 0; i4 < 4; ++i4) {
      int nt = w + i4 * 8;
      int c = nt * 16 + (lane & 15);
#pragma unroll
      for (int r = 0; r < 8; ++r) {
        int m = mt * 16 + r + 8 * hf;
        Hb[m][c] = (__bf16)(gelu_tanh(ag[i4][r]) * av[i4][r]);
      }
    }
  }
  __syncthreads();
  if (t < KTOT) { rsum[t] = 0.f; rssq[t] = 0.f; }
  __syncthreads();
  f32x8 accs[6];
#pragma unroll
  for (int i6 = 0; i6 < 6; ++i6) {
    int tile = w + i6 * 8;
    int mt = tile % 3, nt = tile / 3;
    f32x8 acc = zero8();
    for (int kc = 0; kc < HIDN; kc += 32) {
      bf16x16 a = mk_a([&](int m, int k) { return (float)Hb[m][kc + k]; }, mt * 16);
      bf16x16 b = mk_b([&](int k, int c) { return Wo[(kc + k) * DD + c]; }, nt * 16);
      acc = wmma_bf16(a, b, acc);
    }
    accs[i6] = acc;
    int lane = t & 31, hf = lane >> 4;
#pragma unroll
    for (int r = 0; r < 8; ++r) {
      int m = mt * 16 + r + 8 * hf;
      atomicAdd(&rsum[m], acc[r]);
      atomicAdd(&rssq[m], acc[r] * acc[r]);
    }
  }
  __syncthreads();
  if (t < KTOT) {
    float mu = rsum[t] / DD;
    rmean[t] = mu;
    rrstd[t] = rsqrtf(rssq[t] / DD - mu * mu + 1e-5f);
  }
  __syncthreads();
  float* Fp = fpub + (size_t)n * KTOT * DD;
#pragma unroll
  for (int i6 = 0; i6 < 6; ++i6) {
    int tile = w + i6 * 8;
    int mt = tile % 3, nt = tile / 3;
    st_c(accs[i6],
         [&](int m, int c, float v) {
           float y = (v - rmean[m]) * rrstd[m] * lnps[c] + lnpb[c];
           Fp[m * DD + c] = y;
           Pp[m * DD + c] += y;
         },
         mt * 16, nt * 16);
  }
}

// ---------------------------------------------------------------------------
// K4: msg_in = sum_j mask * sigmoid(P@Wmi)*fpu ; msg_out scatter via atomics
// ---------------------------------------------------------------------------
__global__ void k_msg(const float* pairb, const float* fpub, const int* nbr,
                      const float* mask, const unsigned char* emask, const float* Wmi,
                      const float* Wmo, float* msgin, float* msgout) {
  __shared__ __bf16 Pa[KTOT][DD];   // 24 KB staged pair slab
  __shared__ __bf16 SGb[KTOT][DD];  // 24 KB sigmoid gates (bf16)
  __shared__ float maskf[KTOT];
  int n = blockIdx.x, t = threadIdx.x, w = t >> 5;
  prefetch_w(Wmi, DD * DD, t);
  prefetch_w(Wmo, DD * DD, t);
  if (t < KTOT) {
    bool mk;
    if (t < KNB) {
      int nb = nbr[n * KNB + t];
      mk = (nb != -1) && (mask[nb < 0 ? 0 : nb] > 0.f);
    } else {
      mk = emask[n * EE + (t - KNB)] != 0;
    }
    maskf[t] = mk ? 1.f : 0.f;
  }
  const float* Pp = pairb + (size_t)n * KTOT * DD;
  const float* Fp = fpub + (size_t)n * KTOT * DD;
  for (int i = t; i < KTOT * DD; i += 256) Pa[i / DD][i % DD] = (__bf16)Pp[i];
  __syncthreads();
  for (int tile = w; tile < 48; tile += 8) {
    int mt = tile % 3, nt = tile / 3;
    f32x8 acc = zero8();
    for (int kc = 0; kc < DD; kc += 32) {
      bf16x16 a = mk_a([&](int m, int k) { return (float)Pa[m][kc + k]; }, mt * 16);
      bf16x16 b = mk_b([&](int k, int c) { return Wmi[(kc + k) * DD + c]; }, nt * 16);
      acc = wmma_bf16(a, b, acc);
    }
    st_c(acc,
         [&](int m, int c, float v) { SGb[m][c] = (__bf16)(1.f / (1.f + __expf(-v))); },
         mt * 16, nt * 16);
  }
  __syncthreads();
  {
    float s = 0.f;
    for (int j = 0; j < KTOT; ++j)
      if (maskf[j] > 0.f) s += (float)SGb[j][t] * Fp[j * DD + t];
    msgin[(size_t)n * DD + t] = s;
  }
  __syncthreads();
  for (int tile = w; tile < 32; tile += 8) {  // rows 0..31 only
    int mt = tile % 2, nt = tile / 2;
    f32x8 acc = zero8();
    for (int kc = 0; kc < DD; kc += 32) {
      bf16x16 a = mk_a([&](int m, int k) { return (float)Pa[m][kc + k]; }, mt * 16);
      bf16x16 b = mk_b([&](int k, int c) { return Wmo[(kc + k) * DD + c]; }, nt * 16);
      acc = wmma_bf16(a, b, acc);
    }
    st_c(acc,
         [&](int m, int c, float v) { SGb[m][c] = (__bf16)(1.f / (1.f + __expf(-v))); },
         mt * 16, nt * 16);
  }
  __syncthreads();
  for (int j = 0; j < KNB; ++j) {
    if (maskf[j] > 0.f) {
      int nb = nbr[n * KNB + j];
      if (nb >= 0)
        atomicAdd(&msgout[(size_t)nb * DD + t], (float)SGb[j][t] * Fp[j * DD + t]);
    }
  }
}

// ---------------------------------------------------------------------------
// K5: local gMLP over 16 nodes/workgroup: [local|msg_in|msg_out] 768->512->256
// ---------------------------------------------------------------------------
__global__ void k_gmlp_local(const float* msgin, const float* msgout, const float* Wg,
                             const float* Wv, const float* Wo, const float* lnls,
                             const float* lnlb, float* localb) {
  __shared__ __bf16 Fl[16][F3];    // 24 KB staged lf slab
  __shared__ __bf16 Hb[16][HIDN];  // 16 KB hidden
  __shared__ float rsum[16], rssq[16], rmean[16], rrstd[16];
  int n0 = blockIdx.x * 16, t = threadIdx.x, w = t >> 5;
  prefetch_w(Wg, F3 * HIDN, t);
  prefetch_w(Wv, F3 * HIDN, t);
  for (int i = t; i < 16 * F3; i += 256) {
    int m = i / F3, k = i % F3;
    int node = n0 + m;
    float v;
    if (k < DD) v = localb[(size_t)node * DD + k];
    else if (k < 2 * DD) v = msgin[(size_t)node * DD + (k - DD)];
    else v = msgout[(size_t)node * DD + (k - 2 * DD)];
    Fl[m][k] = (__bf16)v;
  }
  __syncthreads();
  for (int nt = w; nt < 32; nt += 8) {
    f32x8 ag = zero8(), av = zero8();
    for (int kc = 0; kc < F3; kc += 32) {
      bf16x16 a = mk_a([&](int m, int k) { return (float)Fl[m][kc + k]; }, 0);
      bf16x16 bg = mk_b([&](int k, int c) { return Wg[(kc + k) * HIDN + c]; }, nt * 16);
      bf16x16 bv = mk_b([&](int k, int c) { return Wv[(kc + k) * HIDN + c]; }, nt * 16);
      ag = wmma_bf16(a, bg, ag);
      av = wmma_bf16(a, bv, av);
    }
    int lane = t & 31, hf = lane >> 4, c = nt * 16 + (lane & 15);
#pragma unroll
    for (int r = 0; r < 8; ++r)
      Hb[r + 8 * hf][c] = (__bf16)(gelu_tanh(ag[r]) * av[r]);
  }
  __syncthreads();
  if (t < 16) { rsum[t] = 0.f; rssq[t] = 0.f; }
  __syncthreads();
  f32x8 accs[2];
#pragma unroll
  for (int i2 = 0; i2 < 2; ++i2) {
    int nt = w + i2 * 8;
    f32x8 acc = zero8();
    for (int kc = 0; kc < HIDN; kc += 32) {
      bf16x16 a = mk_a([&](int m, int k) { return (float)Hb[m][kc + k]; }, 0);
      bf16x16 b = mk_b([&](int k, int c) { return Wo[(kc + k) * DD + c]; }, nt * 16);
      acc = wmma_bf16(a, b, acc);
    }
    accs[i2] = acc;
    int lane = t & 31, hf = lane >> 4;
#pragma unroll
    for (int r = 0; r < 8; ++r) {
      int m = r + 8 * hf;
      atomicAdd(&rsum[m], acc[r]);
      atomicAdd(&rssq[m], acc[r] * acc[r]);
    }
  }
  __syncthreads();
  if (t < 16) {
    float mu = rsum[t] / DD;
    rmean[t] = mu;
    rrstd[t] = rsqrtf(rssq[t] / DD - mu * mu + 1e-5f);
  }
  __syncthreads();
#pragma unroll
  for (int i2 = 0; i2 < 2; ++i2) {
    int nt = w + i2 * 8;
    st_c(accs[i2],
         [&](int m, int c, float v) {
           float y = (v - rmean[m]) * rrstd[m] * lnls[c] + lnlb[c];
           localb[(size_t)(n0 + m) * DD + c] += y;
         },
         0, nt * 16);
  }
}

// ---------------------------------------------------------------------------
// K6: logits + log_softmax
// ---------------------------------------------------------------------------
__global__ void k_out(const float* localb, const float* Wout, const float* bout,
                      float* out) {
  __shared__ float lg[20];
  __shared__ float mx_s, ls_s;
  int n = blockIdx.x, t = threadIdx.x;
  if (t < 20) {
    float s = bout[t];
    const float* L = localb + (size_t)n * DD;
    for (int k = 0; k < DD; ++k) s += L[k] * Wout[k * 20 + t];
    lg[t] = s;
  }
  __syncthreads();
  if (t == 0) {
    float mx = lg[0];
    for (int c = 1; c < 20; ++c) mx = fmaxf(mx, lg[c]);
    float se = 0.f;
    for (int c = 0; c < 20; ++c) se += __expf(lg[c] - mx);
    mx_s = mx;
    ls_s = logf(se);
  }
  __syncthreads();
  if (t < 20) out[n * 20 + t] = lg[t] - mx_s - ls_s;
}

// ---------------------------------------------------------------------------
// Host launch. Input order assumes JAX sorted-key pytree flatten:
// 0 aa, 1 local, 2 pair, 3 pos, 4 extra_pos, 5 extra_pair, 6 neighbours,
// 7 extra_pair_mask, 8 mask, 9 W_aa_local, 10 W_aa_pair, 11 W_out, 12 b_out,
// layers b at 13+19b: W_extra, W_msg_in, W_msg_out, Wk, Wo_attn, Wq, Wv_attn,
//   gl.Wg, gl.Wo, gl.Wv, gp.Wg, gp.Wo, gp.Wv,
//   ln_attn_b, ln_attn_s, ln_lu_b, ln_lu_s, ln_pu_b, ln_pu_s,
// 51 ln0_b, 52 ln0_s
// ---------------------------------------------------------------------------
extern "C" void kernel_launch(void* const* d_in, const int* in_sizes, int n_in,
                              void* d_out, int out_size, void* d_ws, size_t ws_size,
                              hipStream_t stream) {
  const int* aa = (const int*)d_in[0];
  const float* local = (const float*)d_in[1];
  const float* pair = (const float*)d_in[2];
  const float* extra_pair = (const float*)d_in[5];
  const int* nbr = (const int*)d_in[6];
  const unsigned char* emask = (const unsigned char*)d_in[7];
  const float* mask = (const float*)d_in[8];
  const float* Waal = (const float*)d_in[9];
  const float* Waap = (const float*)d_in[10];
  const float* Wout = (const float*)d_in[11];
  const float* bout = (const float*)d_in[12];
  const float* ln0b = (const float*)d_in[51];
  const float* ln0s = (const float*)d_in[52];

  char* p = (char*)d_ws;
  auto alloc = [&](size_t bytes) -> char* {
    char* r = p;
    p += (bytes + 255) & ~(size_t)255;
    return r;
  };
  float* localb = (float*)alloc((size_t)NN * DD * 4);
  float* pairb = (float*)alloc((size_t)NN * KTOT * DD * 4);
  float* nbb = (float*)alloc((size_t)NN * KTOT * DD * 4);
  float* fpub = (float*)alloc((size_t)NN * KTOT * DD * 4);
  float* msgin = (float*)alloc((size_t)NN * DD * 4);
  float* msgout = (float*)alloc((size_t)NN * DD * 4);
  __bf16* qb = (__bf16*)alloc((size_t)NN * KTOT * DD * 2);
  __bf16* kb = (__bf16*)alloc((size_t)NN * KTOT * DD * 2);
  __bf16* vb = (__bf16*)alloc((size_t)NN * KTOT * DD * 2);

  k_embed<<<NN, 256, 0, stream>>>(aa, local, pair, extra_pair, nbr, Waal, Waap, ln0s,
                                  ln0b, localb, pairb);

  for (int b = 0; b < 2; ++b) {
    int L = 13 + b * 19;
    const float* Wextra = (const float*)d_in[L + 0];
    const float* Wmi = (const float*)d_in[L + 1];
    const float* Wmo = (const float*)d_in[L + 2];
    const float* Wk = (const float*)d_in[L + 3];
    const float* Woa = (const float*)d_in[L + 4];
    const float* Wq = (const float*)d_in[L + 5];
    const float* Wva = (const float*)d_in[L + 6];
    const float* glWg = (const float*)d_in[L + 7];
    const float* glWo = (const float*)d_in[L + 8];
    const float* glWv = (const float*)d_in[L + 9];
    const float* gpWg = (const float*)d_in[L + 10];
    const float* gpWo = (const float*)d_in[L + 11];
    const float* gpWv = (const float*)d_in[L + 12];
    const float* lnab = (const float*)d_in[L + 13];
    const float* lnas = (const float*)d_in[L + 14];
    const float* lnlb = (const float*)d_in[L + 15];
    const float* lnls = (const float*)d_in[L + 16];
    const float* lnpb = (const float*)d_in[L + 17];
    const float* lnps = (const float*)d_in[L + 18];

    k_nbfeat<<<NN, 256, 0, stream>>>(localb, pairb, nbr, Wextra, nbb);
    k_qkv<<<NN, 256, 0, stream>>>(pairb, Wq, Wk, Wva, qb, kb, vb);
    k_attn<<<NN, 256, 0, stream>>>(qb, kb, vb, Woa, lnas, lnab, nbr, mask, emask,
                                   pairb);
    k_gmlp_pair<<<NN, 256, 0, stream>>>(pairb, localb, nbb, gpWg, gpWv, gpWo, lnps,
                                        lnpb, fpub);
    hipMemsetAsync(msgout, 0, (size_t)NN * DD * 4, stream);
    k_msg<<<NN, 256, 0, stream>>>(pairb, fpub, nbr, mask, emask, Wmi, Wmo, msgin,
                                  msgout);
    k_gmlp_local<<<NN / 16, 256, 0, stream>>>(msgin, msgout, glWg, glWv, glWo, lnls,
                                              lnlb, localb);
  }
  k_out<<<NN, 32, 0, stream>>>(localb, Wout, bout, (float*)d_out);
}
